// TigerTokenizer_78082505441553
// MI455X (gfx1250) — compile-verified, matrix operand order checked by reading, not run on hardware
//
#include <hip/hip_runtime.h>
#include <hip/hip_bf16.h>

// ---------------------------------------------------------------------------
// Types for WMMA fragments (CDNA5 / gfx1250, wave32)
// ---------------------------------------------------------------------------
typedef _Float16     half8 __attribute__((ext_vector_type(8)));
typedef _Float16     v16h  __attribute__((ext_vector_type(16)));
typedef float        v8f   __attribute__((ext_vector_type(8)));
typedef float        f4v   __attribute__((ext_vector_type(4)));
typedef unsigned int v4u   __attribute__((ext_vector_type(4)));
typedef int          v8i   __attribute__((ext_vector_type(8)));
typedef int          v4i   __attribute__((ext_vector_type(4)));

// Fragment layout: halves 0..7 = K[kA..kA+7], halves 8..15 = K[kA+16..kA+23]
__device__ __forceinline__ v16h load_frag_f16(const _Float16* p) {
    union { v16h v; half8 h[2]; } u;
    u.h[0] = *(const half8*)(p);
    u.h[1] = *(const half8*)(p + 16);
    return u.v;
}

__device__ __forceinline__ half8 cvt8(const float* __restrict__ p) {
    f4v a = *(const f4v*)(p);
    f4v b = *(const f4v*)(p + 4);
    half8 r;
    r[0] = (_Float16)a[0]; r[1] = (_Float16)a[1];
    r[2] = (_Float16)a[2]; r[3] = (_Float16)a[3];
    r[4] = (_Float16)b[0]; r[5] = (_Float16)b[1];
    r[6] = (_Float16)b[2]; r[7] = (_Float16)b[3];
    return r;
}

__device__ __forceinline__ v16h load_frag_f32(const float* __restrict__ p) {
    union { v16h v; half8 h[2]; } u;
    u.h[0] = cvt8(p);
    u.h[1] = cvt8(p + 16);
    return u.v;
}

__device__ __forceinline__ v8f wmma_f16(v16h a, v16h b, v8f c) {
    return __builtin_amdgcn_wmma_f32_16x16x32_f16(false, a, false, b,
                                                  (short)0, c, false, false);
}

// ---------------------------------------------------------------------------
// TDM: stage a contiguous [nrows x rowDw dwords] tile from global into LDS.
// D# per CDNA5 ISA (2-D, data_size = 4 bytes, groups 2/3 = 0)
// This toolchain's builtin is the 6-arg (clang-23 / therock) form:
//   (uint32x4 g0, int32x8 g1, int32x4, int32x4, int32x8, i32 cpol)
// ---------------------------------------------------------------------------
__device__ __forceinline__ void tdm_load_tile(const void* gptr, unsigned ldsAddr,
                                              unsigned rowDw, unsigned nrows) {
#if __has_builtin(__builtin_amdgcn_tensor_load_to_lds)
    unsigned long long ga = (unsigned long long)(uintptr_t)gptr;
    v4u g0;
    g0[0] = 1u;                                   // count=1, user descriptor
    g0[1] = ldsAddr;                              // lds_addr
    g0[2] = (unsigned)ga;                         // global_addr[31:0]
    g0[3] = ((unsigned)(ga >> 32) & 0x01FFFFFFu)  // global_addr[56:32]
            | 0x80000000u;                        // type=2 ("image")
    v8i g1;
    g1[0] = (int)(2u << 16);                      // wg_mask=0, data_size=2 (4B)
    g1[1] = (int)((rowDw & 0xFFFFu) << 16);       // tensor_dim0[15:0]
    g1[2] = (int)(((rowDw >> 16) & 0xFFFFu) |     // tensor_dim0[31:16]
                  ((nrows & 0xFFFFu) << 16));     // tensor_dim1[15:0]
    g1[3] = (int)(((nrows >> 16) & 0xFFFFu) |     // tensor_dim1[31:16]
                  ((rowDw & 0xFFFFu) << 16));     // tile_dim0
    g1[4] = (int)(nrows & 0xFFFFu);               // tile_dim1 (tile_dim2=0)
    g1[5] = (int)rowDw;                           // tensor_dim0_stride[31:0]
    g1[6] = 0;                                    // stride[47:32], dim1_stride lo
    g1[7] = 0;
    v4i gz4 = {0, 0, 0, 0};
    v8i gz8 = {0, 0, 0, 0, 0, 0, 0, 0};
    __builtin_amdgcn_tensor_load_to_lds(g0, g1, gz4, gz4, gz8, 0);
#endif
}

// ---------------------------------------------------------------------------
// GEMM: Y[m,n] = act( sum_k X[m,k] * W[n,k] + bias[n] ), X/W f16, acc f32.
// Block = NG*MT waves. The block's MT 16-row A-stripes are DMA'd to LDS once
// (TDM) and shared by the NG n-group waves (each owning 64 output columns).
// ---------------------------------------------------------------------------
template <int NG, int MT, bool RELU, typename OutT>
__global__ __launch_bounds__(32 * NG * MT) void gemm_lds(
    const _Float16* __restrict__ X, const _Float16* __restrict__ W,
    const float* __restrict__ bias, OutT* __restrict__ Y,
    int Cin, int Cout) {
    extern __shared__ _Float16 sA[];              // [MT*16][Cin]

    const size_t rowBase = (size_t)blockIdx.x * (MT * 16) * Cin;

#if __has_builtin(__builtin_amdgcn_tensor_load_to_lds)
    if (threadIdx.x < 32) {
        if (threadIdx.x == 0)
            tdm_load_tile(X + rowBase, (unsigned)(uintptr_t)sA,
                          (unsigned)(Cin >> 1), (unsigned)(MT * 16));
        __builtin_amdgcn_s_wait_tensorcnt(0);
    }
#else
    const int chunks = MT * 16 * Cin / 8;
    for (int i = threadIdx.x; i < chunks; i += blockDim.x)
        ((half8*)sA)[i] = ((const half8*)(X + rowBase))[i];
#endif
    __syncthreads();

    const int wave    = threadIdx.x >> 5;
    const int mSub    = wave / NG;
    const int nBase   = (wave % NG) << 6;
    const int lane    = threadIdx.x & 31;
    const int halfSel = lane >> 4;
    const int lrow    = lane & 15;

    const _Float16* arow = sA + (size_t)(mSub * 16 + lrow) * Cin;

    v8f acc[4] = {};
    for (int k0 = 0; k0 < Cin; k0 += 32) {
        const int kA = k0 + halfSel * 8;
        v16h a = load_frag_f16(arow + kA);        // ds_load_b128 x2
#pragma unroll
        for (int t = 0; t < 4; ++t) {
            const int n = nBase + t * 16 + lrow;
            v16h b = load_frag_f16(W + (size_t)n * Cin + kA);
            acc[t] = wmma_f16(a, b, acc[t]);
        }
    }
#pragma unroll
    for (int t = 0; t < 4; ++t) {
        const int n    = nBase + t * 16 + lrow;
        const float bv = bias[n];
#pragma unroll
        for (int r = 0; r < 8; ++r) {
            const int m = blockIdx.x * (MT * 16) + mSub * 16 + halfSel * 8 + r;
            float v = acc[t][r] + bv;
            if (RELU) v = v > 0.0f ? v : 0.0f;
            Y[(size_t)m * Cout + n] = (OutT)v;
        }
    }
}

// ---------------------------------------------------------------------------
// One residual-VQ step: scores via WMMA, argmin (first-min tie-break),
// in-place residual update, deterministic per-block ||res||^2 partials.
// Block = 8 waves, each wave owns 16 rows -> 128 rows/block.
// ---------------------------------------------------------------------------
__global__ __launch_bounds__(256) void vq_step(
    float* __restrict__ res, const _Float16* __restrict__ cb_h,
    const float* __restrict__ cb_f, const float* __restrict__ cbnorm,
    float* __restrict__ idx_out, int l, float* __restrict__ partial) {
    __shared__ int   sIdx[8][16];
    __shared__ float sRed[256];

    const int waveId  = threadIdx.x >> 5;
    const int lane    = threadIdx.x & 31;
    const int halfSel = lane >> 4;
    const int lrow    = lane & 15;
    const int mBase   = (blockIdx.x * 8 + waveId) * 16;

    const float* rrow = res + (size_t)(mBase + lrow) * 128;
    v16h afr[4];
#pragma unroll
    for (int ks = 0; ks < 4; ++ks)
        afr[ks] = load_frag_f32(rrow + ks * 32 + halfSel * 8);

    float minv[8];
    int   mini[8];
#pragma unroll
    for (int r = 0; r < 8; ++r) { minv[r] = 3.4e38f; mini[r] = 0x7fffffff; }

    for (int nt = 0; nt < 16; ++nt) {
        const int n = nt * 16 + lrow;
        v8f acc = {};
#pragma unroll
        for (int ks = 0; ks < 4; ++ks) {
            v16h b = load_frag_f16(cb_h + (size_t)n * 128 + ks * 32 + halfSel * 8);
            acc = wmma_f16(afr[ks], b, acc);
        }
        const float cn = cbnorm[n];
#pragma unroll
        for (int r = 0; r < 8; ++r) {
            const float score = cn - 2.0f * acc[r];
            if (score < minv[r]) { minv[r] = score; mini[r] = n; }
        }
    }

#pragma unroll
    for (int r = 0; r < 8; ++r) {
        float v = minv[r];
        int   i = mini[r];
#pragma unroll
        for (int mask = 1; mask < 16; mask <<= 1) {
            float ov = __shfl_xor(v, mask, 32);
            int   oi = __shfl_xor(i, mask, 32);
            if (ov < v || (ov == v && oi < i)) { v = ov; i = oi; }
        }
        if (lrow == 0) sIdx[waveId][halfSel * 8 + r] = i;
    }
    __syncthreads();

    float sq = 0.0f;
    for (int row = 0; row < 16; ++row) {
        const int idx  = sIdx[waveId][row];
        const size_t g = (size_t)(mBase + row);
        if (lane == 0) idx_out[g * 3 + l] = (float)idx;
        const float* q = cb_f + (size_t)idx * 128;
        float* rr = res + g * 128;
#pragma unroll
        for (int d = lane; d < 128; d += 32) {
            const float nv = rr[d] - q[d];
            rr[d] = nv;
            sq += nv * nv;
        }
    }

    sRed[threadIdx.x] = sq;
    __syncthreads();
    for (int s = 128; s > 0; s >>= 1) {
        if (threadIdx.x < (unsigned)s) sRed[threadIdx.x] += sRed[threadIdx.x + s];
        __syncthreads();
    }
    if (threadIdx.x == 0) partial[blockIdx.x] = sRed[0];
}

// ---------------------------------------------------------------------------
// Small utility kernels
// ---------------------------------------------------------------------------
__global__ void f32_to_f16_v8(const float* __restrict__ src,
                              _Float16* __restrict__ dst, int n8) {
    const int i = blockIdx.x * blockDim.x + threadIdx.x;
    if (i < n8) *((half8*)dst + i) = cvt8(src + (size_t)i * 8);
}

__global__ void cb_norms(const float* __restrict__ cb, float* __restrict__ out, int n) {
    const int i = blockIdx.x * blockDim.x + threadIdx.x;
    if (i < n) {
        const float* r = cb + (size_t)i * 128;
        float s = 0.0f;
        for (int d = 0; d < 128; ++d) s += r[d] * r[d];
        out[i] = s;
    }
}

__global__ void make_xq(const float* __restrict__ z, const float* __restrict__ res,
                        _Float16* __restrict__ xq, int n) {
    const int i = blockIdx.x * blockDim.x + threadIdx.x;
    if (i < n) xq[i] = (_Float16)(z[i] - res[i]);
}

__global__ void finalize_loss(const float* __restrict__ partial, float* __restrict__ out,
                              int n, float scale) {
    __shared__ float s[256];
    float acc = 0.0f;
    for (int i = threadIdx.x; i < n; i += 256) acc += partial[i];
    s[threadIdx.x] = acc;
    __syncthreads();
    for (int k = 128; k > 0; k >>= 1) {
        if (threadIdx.x < (unsigned)k) s[threadIdx.x] += s[threadIdx.x + k];
        __syncthreads();
    }
    if (threadIdx.x == 0) *out = s[0] * scale;
}

// ---------------------------------------------------------------------------
// Launcher
// ---------------------------------------------------------------------------
extern "C" void kernel_launch(void* const* d_in, const int* in_sizes, int n_in,
                              void* d_out, int out_size, void* d_ws, size_t ws_size,
                              hipStream_t stream) {
    const int N = in_sizes[0] / 768;  // 131072

    const float* emb    = (const float*)d_in[0];
    const float* enc_w0 = (const float*)d_in[1];
    const float* enc_b0 = (const float*)d_in[2];
    const float* dec_w0 = (const float*)d_in[3];
    const float* dec_b0 = (const float*)d_in[4];
    const float* enc_w1 = (const float*)d_in[5];
    const float* enc_b1 = (const float*)d_in[6];
    const float* dec_w1 = (const float*)d_in[7];
    const float* dec_b1 = (const float*)d_in[8];
    const float* enc_w2 = (const float*)d_in[9];
    const float* enc_b2 = (const float*)d_in[10];
    const float* dec_w2 = (const float*)d_in[11];
    const float* dec_b2 = (const float*)d_in[12];
    const float* cb     = (const float*)d_in[13];  // [3,256,128]

    float* out      = (float*)d_out;
    float* recon    = out;                          // [N,768]
    float* idx_out  = out + (size_t)N * 768;        // [N,3]
    float* loss_out = idx_out + (size_t)N * 3;      // [1]

    char*  ws  = (char*)d_ws;
    size_t off = 0;
    auto alloc = [&](size_t bytes) -> char* {
        off = (off + 255) & ~(size_t)255;
        char* p = ws + off;
        off += bytes;
        return p;
    };

    _Float16* enc_w0_h = (_Float16*)alloc((size_t)512 * 768 * 2);
    _Float16* enc_w1_h = (_Float16*)alloc((size_t)256 * 512 * 2);
    _Float16* enc_w2_h = (_Float16*)alloc((size_t)128 * 256 * 2);
    _Float16* dec_w0_h = (_Float16*)alloc((size_t)256 * 128 * 2);
    _Float16* dec_w1_h = (_Float16*)alloc((size_t)512 * 256 * 2);
    _Float16* dec_w2_h = (_Float16*)alloc((size_t)768 * 512 * 2);
    _Float16* cb_h     = (_Float16*)alloc((size_t)3 * 256 * 128 * 2);
    float*    cbnorm   = (float*)   alloc((size_t)3 * 256 * 4);
    _Float16* emb_h    = (_Float16*)alloc((size_t)N * 768 * 2);
    _Float16* h1_h     = (_Float16*)alloc((size_t)N * 512 * 2);
    _Float16* h2_h     = (_Float16*)alloc((size_t)N * 256 * 2);
    float*    z        = (float*)   alloc((size_t)N * 128 * 4);
    float*    res      = (float*)   alloc((size_t)N * 128 * 4);
    _Float16* xq_h     = (_Float16*)alloc((size_t)N * 128 * 2);
    _Float16* d1_h     = (_Float16*)alloc((size_t)N * 256 * 2);
    _Float16* d2_h     = (_Float16*)alloc((size_t)N * 512 * 2);
    float*    partial  = (float*)   alloc((size_t)3 * (N / 128) * 4);

    auto cvt = [&](const float* s, _Float16* d, int n) {
        f32_to_f16_v8<<<(n / 8 + 255) / 256, 256, 0, stream>>>(s, d, n / 8);
    };
    cvt(enc_w0, enc_w0_h, 512 * 768);
    cvt(enc_w1, enc_w1_h, 256 * 512);
    cvt(enc_w2, enc_w2_h, 128 * 256);
    cvt(dec_w0, dec_w0_h, 256 * 128);
    cvt(dec_w1, dec_w1_h, 512 * 256);
    cvt(dec_w2, dec_w2_h, 768 * 512);
    cvt(cb, cb_h, 3 * 256 * 128);
    cvt(emb, emb_h, N * 768);
    cb_norms<<<3, 256, 0, stream>>>(cb, cbnorm, 3 * 256);

    // Encoder: block stages MT*16 rows of A in LDS (TDM) shared by NG n-groups
    gemm_lds<8, 1, true, _Float16>
        <<<N / 16, 256, 16 * 768 * 2, stream>>>(emb_h, enc_w0_h, enc_b0, h1_h, 768, 512);
    gemm_lds<4, 2, true, _Float16>
        <<<N / 32, 256, 32 * 512 * 2, stream>>>(h1_h, enc_w1_h, enc_b1, h2_h, 512, 256);
    gemm_lds<2, 4, false, float>
        <<<N / 64, 256, 64 * 256 * 2, stream>>>(h2_h, enc_w2_h, enc_b2, z, 256, 128);

    (void)hipMemcpyAsync(res, z, (size_t)N * 128 * 4, hipMemcpyDeviceToDevice, stream);

    // Residual VQ: 3 sequential steps
    const int vqBlocks = N / 128;
    for (int l = 0; l < 3; ++l) {
        vq_step<<<vqBlocks, 256, 0, stream>>>(
            res, cb_h + (size_t)l * 256 * 128, cb + (size_t)l * 256 * 128,
            cbnorm + l * 256, idx_out, l, partial + (size_t)l * vqBlocks);
    }
    finalize_loss<<<1, 256, 0, stream>>>(partial, loss_out, 3 * vqBlocks,
                                         1.25f / ((float)N * 128.0f));

    // Straight-through forward: x_q = z - res_final
    make_xq<<<(N * 128 + 255) / 256, 256, 0, stream>>>(z, res, xq_h, N * 128);

    // Decoder
    gemm_lds<4, 2, true, _Float16>
        <<<N / 32, 256, 32 * 128 * 2, stream>>>(xq_h, dec_w0_h, dec_b0, d1_h, 128, 256);
    gemm_lds<8, 1, true, _Float16>
        <<<N / 16, 256, 16 * 256 * 2, stream>>>(d1_h, dec_w1_h, dec_b1, d2_h, 256, 512);
    gemm_lds<12, 1, false, float>
        <<<N / 16, 384, 16 * 512 * 2, stream>>>(d2_h, dec_w2_h, dec_b2, recon, 512, 768);
}
